// GAERecurrent_53008486367983
// MI455X (gfx1250) — matrix operand order repaired and checked by reading, not compiled, not run on hardware
//
#include <hip/hip_runtime.h>
#include <hip/hip_bf16.h>

#define NN      100000
#define NFEAT   256
#define NHID    128
#define EDGES   3200000
#define NPAIR   200000
#define STEPS   5

typedef __attribute__((ext_vector_type(16))) __bf16 v16bf;
typedef __attribute__((ext_vector_type(8)))  float  v8f;

union Frag16 { v16bf v; uint4 u4[2]; };

__device__ __forceinline__ unsigned short f2bf(float f) {
    unsigned x = __float_as_uint(f);
    unsigned r = (x + 0x7FFFu + ((x >> 16) & 1u)) >> 16;   // round-to-nearest-even
    return (unsigned short)r;
}
__device__ __forceinline__ float sigmoidf_(float x) { return 1.0f / (1.0f + __expf(-x)); }

// gfx1250 async global->LDS copy of 16 bytes (per-lane), tracked by ASYNCcnt
__device__ __forceinline__ void async_g2l_b128(unsigned lds_byte_addr, const void* gsrc) {
    unsigned long long ga = (unsigned long long)gsrc;
    asm volatile("global_load_async_to_lds_b128 %0, %1, off"
                 :: "v"(lds_byte_addr), "v"(ga) : "memory");
}
__device__ __forceinline__ void wait_asynccnt0() {
    asm volatile("s_wait_asynccnt 0x0" ::: "memory");
}

// ---------------------------------------------------------------------------
// f32 -> bf16 convert (grid-stride)
// ---------------------------------------------------------------------------
__global__ void conv_bf16_kernel(const float* __restrict__ x, unsigned short* __restrict__ y, long n) {
    long i = (long)blockIdx.x * blockDim.x + threadIdx.x;
    long stride = (long)gridDim.x * blockDim.x;
    for (; i < n; i += stride) y[i] = f2bf(x[i]);
}

// transpose + convert weight: W [K, Ncols] f32 row-major -> Wt [Ncols, K] bf16
__global__ void wtrans_kernel(const float* __restrict__ W, unsigned short* __restrict__ Wt,
                              int K, int Ncols) {
    int idx = blockIdx.x * blockDim.x + threadIdx.x;
    if (idx < K * Ncols) {
        int k = idx / Ncols, n = idx % Ncols;
        Wt[n * K + k] = f2bf(W[idx]);
    }
}

__global__ void zero_kernel(float* __restrict__ p, long n) {
    long i = (long)blockIdx.x * blockDim.x + threadIdx.x;
    long stride = (long)gridDim.x * blockDim.x;
    for (; i < n; i += stride) p[i] = 0.0f;
}

// ---------------------------------------------------------------------------
// WMMA GEMM: C[M,128] = A[M,K](bf16) @ W[K,128] (given as Wt[128,K] bf16) + bias
// block = 256 threads (8 waves) -> 32 rows x 128 cols.
// wave w: row-tile = w>>2 (16 rows), col-quarter = w&3 (32 cols = 2 acc tiles).
// Full Wt staged in LDS via async global->LDS (ASYNCcnt), B frags from ds_load.
// ---------------------------------------------------------------------------
__global__ __launch_bounds__(256) void gemm_bf16_wmma(
    const unsigned short* __restrict__ A,
    const unsigned short* __restrict__ Wt,
    const float* __restrict__ bias,
    float* __restrict__ C,
    int M, int K, int relu)
{
    extern __shared__ unsigned short smem[];   // [128 * K] bf16 weight tile

    const int tid  = threadIdx.x;
    const int wave = tid >> 5;
    const int lane = tid & 31;
    const int l15  = lane & 15;
    const int half = lane >> 4;
    const int rt   = wave >> 2;                 // 0..1
    const int cq   = wave & 3;                  // 0..3
    const int row_base = blockIdx.x * 32 + rt * 16;
    const int n0   = cq * 32 + l15;
    const int n1   = n0 + 16;
    const int row  = row_base + l15;

    // ---- stage Wt[128,K] into LDS with async copies --------------------------
    const unsigned lds0 = (unsigned)(unsigned long long)(&smem[0]);
    const int chunks = (NHID * K) >> 3;          // 16-byte chunks
    for (int c = tid; c < chunks; c += 256)
        async_g2l_b128(lds0 + c * 16, Wt + c * 8);
    wait_asynccnt0();
    __syncthreads();

    v8f acc0, acc1;
    const float bv0 = bias ? bias[n0] : 0.0f;
    const float bv1 = bias ? bias[n1] : 0.0f;
#pragma unroll
    for (int j = 0; j < 8; ++j) { acc0[j] = bv0; acc1[j] = bv1; }

    const uint4* Arow = reinterpret_cast<const uint4*>(A + (size_t)row * K);
    const uint4* B0   = reinterpret_cast<const uint4*>(smem + (size_t)n0 * K);
    const uint4* B1   = reinterpret_cast<const uint4*>(smem + (size_t)n1 * K);

    for (int kk = 0; kk < K; kk += 32) {
        Frag16 a, b0, b1;
        // A 16x32 bf16 layout: lanes<16 -> K {kk..kk+7, kk+16..kk+23}; lanes>=16 -> +8
        a.u4[0] = Arow[(kk + 8 * half) >> 3];
        a.u4[1] = Arow[(kk + 16 + 8 * half) >> 3];
        // B 32x16 bf16 layout: lanes<16 -> K {kk..kk+7, kk+16..kk+23}; lanes>=16 -> +8
        b0.u4[0] = B0[(kk + 8 * half) >> 3];
        b0.u4[1] = B0[(kk + 16 + 8 * half) >> 3];
        b1.u4[0] = B1[(kk + 8 * half) >> 3];
        b1.u4[1] = B1[(kk + 16 + 8 * half) >> 3];
        acc0 = __builtin_amdgcn_wmma_f32_16x16x32_bf16(
            false, a.v, false, b0.v, (short)0, acc0, false, false);
        acc1 = __builtin_amdgcn_wmma_f32_16x16x32_bf16(
            false, a.v, false, b1.v, (short)0, acc1, false, false);
    }

#pragma unroll
    for (int j = 0; j < 8; ++j) {
        float v0 = acc0[j], v1 = acc1[j];
        if (relu) { v0 = v0 > 0.0f ? v0 : 0.0f; v1 = v1 > 0.0f ? v1 : 0.0f; }
        const size_t r = (size_t)(row_base + j + 8 * half) * NHID;
        C[r + n0] = v0;
        C[r + n1] = v1;
    }
}

// ---------------------------------------------------------------------------
// SpMM accumulate: out[row,:] += w * hnb[col,:], rows sorted -> run-length
// register accumulation, atomic flush at run boundaries. 128 thr = 1 feature each.
// ---------------------------------------------------------------------------
__global__ __launch_bounds__(128) void spmm_add_kernel(
    const int* __restrict__ rows, const int* __restrict__ cols,
    const float* __restrict__ w, const float* __restrict__ hnb,
    float* __restrict__ out, int E)
{
    const int f  = threadIdx.x;
    const int e0 = blockIdx.x * 16;
    const int e1 = e0 + 16 < E ? e0 + 16 : E;
    int cur = rows[e0];
    float acc = 0.0f;
    for (int e = e0; e < e1; ++e) {
        int r = rows[e];
        if (r != cur) {
            atomicAdd(&out[(size_t)cur * NHID + f], acc);
            acc = 0.0f; cur = r;
        }
        acc += hnb[(size_t)cols[e] * NHID + f] * w[e];
    }
    atomicAdd(&out[(size_t)cur * NHID + f], acc);
}

// ---------------------------------------------------------------------------
// GRU gate update: g_* already hold self-proj + neighbor aggregation
// ---------------------------------------------------------------------------
__global__ void gru_update_kernel(
    const float* __restrict__ in_r, const float* __restrict__ in_i, const float* __restrict__ in_n,
    const float* __restrict__ gr,   const float* __restrict__ gi,   const float* __restrict__ gh,
    float* __restrict__ h, long n)
{
    long i = (long)blockIdx.x * blockDim.x + threadIdx.x;
    long stride = (long)gridDim.x * blockDim.x;
    for (; i < n; i += stride) {
        float r  = sigmoidf_(in_r[i] + gr[i]);
        float ii = sigmoidf_(in_i[i] + gi[i]);
        float nn = tanhf(in_n[i] + r * gh[i]);
        h[i] = (1.0f - ii) * nn + ii * h[i];
    }
}

// ---------------------------------------------------------------------------
// Torch-style LayerNorm (unbiased std, eps on std), 1 wave per node
// writes z (f32) and z_bf16
// ---------------------------------------------------------------------------
__global__ __launch_bounds__(256) void layernorm_kernel(
    const float* __restrict__ x, const float* __restrict__ gamma, const float* __restrict__ beta,
    float* __restrict__ z, unsigned short* __restrict__ zbf, int nNodes)
{
    int wave = threadIdx.x >> 5, lane = threadIdx.x & 31;
    int node = blockIdx.x * 8 + wave;
    if (node >= nNodes) return;
    const float* xr = x + (size_t)node * NHID;
    float v[4], s = 0.0f, s2 = 0.0f;
#pragma unroll
    for (int j = 0; j < 4; ++j) { v[j] = xr[lane + 32 * j]; s += v[j]; s2 += v[j] * v[j]; }
#pragma unroll
    for (int m = 16; m >= 1; m >>= 1) { s += __shfl_xor(s, m, 32); s2 += __shfl_xor(s2, m, 32); }
    float mean = s * (1.0f / NHID);
    float var  = (s2 - s * mean) * (1.0f / (NHID - 1));   // unbiased
    if (var < 0.0f) var = 0.0f;
    float inv = 1.0f / (sqrtf(var) + 1e-6f);
#pragma unroll
    for (int j = 0; j < 4; ++j) {
        int c = lane + 32 * j;
        float zz = gamma[c] * (v[j] - mean) * inv + beta[c];
        z[(size_t)node * NHID + c]   = zz;
        zbf[(size_t)node * NHID + c] = f2bf(zz);
    }
}

// gather pair features: Ag[p, 0:128]=z[x_idx[p]], Ag[p,128:256]=z[y_idx[p]] (bf16)
__global__ void pair_gather_kernel(const unsigned short* __restrict__ zbf,
                                   const int* __restrict__ xi, const int* __restrict__ yi,
                                   unsigned short* __restrict__ Ag, int P)
{
    int t = blockIdx.x * 256 + threadIdx.x;
    if (t < P * 256) {
        int p = t >> 8, c = t & 255;
        int node = (c < NHID) ? xi[p] : yi[p];
        Ag[(size_t)p * 256 + c] = zbf[(size_t)node * NHID + (c & 127)];
    }
}

// final fc2 + sigmoid: out[p] = sigmoid(dot(d1[p,:], Wd2) + bd2); 1 wave / pair
__global__ __launch_bounds__(256) void fc2_sigmoid_kernel(
    const float* __restrict__ d1, const float* __restrict__ Wd2, const float* __restrict__ bd2,
    float* __restrict__ out, int P)
{
    int wave = threadIdx.x >> 5, lane = threadIdx.x & 31;
    int p = blockIdx.x * 8 + wave;
    if (p >= P) return;
    const float* r = d1 + (size_t)p * NHID;
    float s = 0.0f;
#pragma unroll
    for (int j = 0; j < 4; ++j) s += r[lane + 32 * j] * Wd2[lane + 32 * j];
#pragma unroll
    for (int m = 16; m >= 1; m >>= 1) s += __shfl_xor(s, m, 32);
    if (lane == 0) out[p] = sigmoidf_(s + bd2[0]);
}

// ---------------------------------------------------------------------------
extern "C" void kernel_launch(void* const* d_in, const int* in_sizes, int n_in,
                              void* d_out, int out_size, void* d_ws, size_t ws_size,
                              hipStream_t stream) {
    (void)in_sizes; (void)n_in; (void)out_size; (void)ws_size;

    const float* inputs   = (const float*)d_in[0];
    const int*   edge_row = (const int*)d_in[1];
    const int*   edge_col = (const int*)d_in[2];
    const float* edge_w   = (const float*)d_in[3];
    const int*   x_idx    = (const int*)d_in[4];
    const int*   y_idx    = (const int*)d_in[5];
    const float* Wr_in = (const float*)d_in[6];  const float* br_in = (const float*)d_in[7];
    const float* Wi_in = (const float*)d_in[8];  const float* bi_in = (const float*)d_in[9];
    const float* Wn_in = (const float*)d_in[10]; const float* bn_in = (const float*)d_in[11];
    const float* Hr_self = (const float*)d_in[12]; const float* Hr_nb = (const float*)d_in[13];
    const float* Hi_self = (const float*)d_in[14]; const float* Hi_nb = (const float*)d_in[15];
    const float* Hh_self = (const float*)d_in[16]; const float* Hh_nb = (const float*)d_in[17];
    const float* W1 = (const float*)d_in[18]; const float* b1 = (const float*)d_in[19];
    const float* W2 = (const float*)d_in[20]; const float* b2 = (const float*)d_in[21];
    const float* W3 = (const float*)d_in[22]; const float* b3 = (const float*)d_in[23];
    const float* gamma = (const float*)d_in[24]; const float* beta = (const float*)d_in[25];
    const float* Wd1 = (const float*)d_in[26]; const float* bd1 = (const float*)d_in[27];
    const float* Wd2 = (const float*)d_in[28]; const float* bd2 = (const float*)d_in[29];

    float* out_d = (float*)d_out;            // [P]
    float* out_z = (float*)d_out + NPAIR;    // [N,128]

    // --- deterministic bump allocator over d_ws -----------------------------
    char* cur = (char*)d_ws;
    auto alloc = [&](size_t bytes) -> void* {
        void* p = (void*)cur;
        cur += (bytes + 255) & ~(size_t)255;
        return p;
    };
    // transposed bf16 weights
    unsigned short* WrT = (unsigned short*)alloc((size_t)NFEAT * NHID * 2);
    unsigned short* WiT = (unsigned short*)alloc((size_t)NFEAT * NHID * 2);
    unsigned short* WnT = (unsigned short*)alloc((size_t)NFEAT * NHID * 2);
    unsigned short* Wd1T = (unsigned short*)alloc((size_t)(2 * NHID) * NHID * 2);
    unsigned short* HrsT = (unsigned short*)alloc((size_t)NHID * NHID * 2);
    unsigned short* HrnT = (unsigned short*)alloc((size_t)NHID * NHID * 2);
    unsigned short* HisT = (unsigned short*)alloc((size_t)NHID * NHID * 2);
    unsigned short* HinT = (unsigned short*)alloc((size_t)NHID * NHID * 2);
    unsigned short* HhsT = (unsigned short*)alloc((size_t)NHID * NHID * 2);
    unsigned short* HhnT = (unsigned short*)alloc((size_t)NHID * NHID * 2);
    unsigned short* W1T = (unsigned short*)alloc((size_t)NHID * NHID * 2);
    unsigned short* W2T = (unsigned short*)alloc((size_t)NHID * NHID * 2);
    unsigned short* W3T = (unsigned short*)alloc((size_t)NHID * NHID * 2);
    // big activation buffers
    unsigned short* in_bf = (unsigned short*)alloc((size_t)NN * NFEAT * 2);
    float* in_r = (float*)alloc((size_t)NN * NHID * 4);   // + in_i contiguous -> reused as Ag
    float* in_i = (float*)alloc((size_t)NN * NHID * 4);
    float* in_n = (float*)alloc((size_t)NN * NHID * 4);   // + h contiguous -> reused as d1
    float* h    = (float*)alloc((size_t)NN * NHID * 4);
    unsigned short* hbf = (unsigned short*)alloc((size_t)NN * NHID * 2);
    float* g_r = (float*)alloc((size_t)NN * NHID * 4);
    float* g_i = (float*)alloc((size_t)NN * NHID * 4);
    float* g_h = (float*)alloc((size_t)NN * NHID * 4);
    float* hnb = (float*)alloc((size_t)NN * NHID * 4);
    unsigned short* xbf = (unsigned short*)alloc((size_t)NN * NHID * 2);
    // decoder reuses loop-phase buffers (all free by then)
    unsigned short* Ag  = (unsigned short*)in_r;          // P*256 bf16 over in_r+in_i
    float*          d1  = (float*)in_n;                   // P*128 f32 over in_n+h
    unsigned short* zbf = (unsigned short*)hnb;           // 25.6MB over 51.2MB

    const long NH = (long)NN * NHID;
    const int GRID_GEMM_N = NN / 32;      // 3125 (32 rows per block)
    const int GRID_GEMM_P = NPAIR / 32;   // 6250
    const size_t LDS128 = (size_t)NHID * NHID * 2;   // 32 KB
    const size_t LDS256 = (size_t)NHID * NFEAT * 2;  // 64 KB
    const int EW_GRID = (int)(NH + 255) / 256 < 65535 ? (int)((NH + 255) / 256) : 65535;

    // 1) convert inputs + weights
    conv_bf16_kernel<<<4096, 256, 0, stream>>>(inputs, in_bf, (long)NN * NFEAT);
    wtrans_kernel<<<(NFEAT * NHID + 255) / 256, 256, 0, stream>>>(Wr_in, WrT, NFEAT, NHID);
    wtrans_kernel<<<(NFEAT * NHID + 255) / 256, 256, 0, stream>>>(Wi_in, WiT, NFEAT, NHID);
    wtrans_kernel<<<(NFEAT * NHID + 255) / 256, 256, 0, stream>>>(Wn_in, WnT, NFEAT, NHID);
    wtrans_kernel<<<(2 * NHID * NHID + 255) / 256, 256, 0, stream>>>(Wd1, Wd1T, 2 * NHID, NHID);
    wtrans_kernel<<<(NHID * NHID + 255) / 256, 256, 0, stream>>>(Hr_self, HrsT, NHID, NHID);
    wtrans_kernel<<<(NHID * NHID + 255) / 256, 256, 0, stream>>>(Hr_nb,   HrnT, NHID, NHID);
    wtrans_kernel<<<(NHID * NHID + 255) / 256, 256, 0, stream>>>(Hi_self, HisT, NHID, NHID);
    wtrans_kernel<<<(NHID * NHID + 255) / 256, 256, 0, stream>>>(Hi_nb,   HinT, NHID, NHID);
    wtrans_kernel<<<(NHID * NHID + 255) / 256, 256, 0, stream>>>(Hh_self, HhsT, NHID, NHID);
    wtrans_kernel<<<(NHID * NHID + 255) / 256, 256, 0, stream>>>(Hh_nb,   HhnT, NHID, NHID);
    wtrans_kernel<<<(NHID * NHID + 255) / 256, 256, 0, stream>>>(W1, W1T, NHID, NHID);
    wtrans_kernel<<<(NHID * NHID + 255) / 256, 256, 0, stream>>>(W2, W2T, NHID, NHID);
    wtrans_kernel<<<(NHID * NHID + 255) / 256, 256, 0, stream>>>(W3, W3T, NHID, NHID);

    // 2) loop-invariant input projections
    gemm_bf16_wmma<<<GRID_GEMM_N, 256, LDS256, stream>>>(in_bf, WrT, br_in, in_r, NN, NFEAT, 0);
    gemm_bf16_wmma<<<GRID_GEMM_N, 256, LDS256, stream>>>(in_bf, WiT, bi_in, in_i, NN, NFEAT, 0);
    gemm_bf16_wmma<<<GRID_GEMM_N, 256, LDS256, stream>>>(in_bf, WnT, bn_in, in_n, NN, NFEAT, 0);

    // 3) h = 0
    zero_kernel<<<EW_GRID, 256, 0, stream>>>(h, NH);

    // 4) GRU + GCN recurrence
    for (int step = 0; step < STEPS; ++step) {
        conv_bf16_kernel<<<4096, 256, 0, stream>>>(h, hbf, NH);
        // gate r
        gemm_bf16_wmma<<<GRID_GEMM_N, 256, LDS128, stream>>>(hbf, HrsT, nullptr, g_r, NN, NHID, 0);
        gemm_bf16_wmma<<<GRID_GEMM_N, 256, LDS128, stream>>>(hbf, HrnT, nullptr, hnb, NN, NHID, 0);
        spmm_add_kernel<<<EDGES / 16, 128, 0, stream>>>(edge_row, edge_col, edge_w, hnb, g_r, EDGES);
        // gate i
        gemm_bf16_wmma<<<GRID_GEMM_N, 256, LDS128, stream>>>(hbf, HisT, nullptr, g_i, NN, NHID, 0);
        gemm_bf16_wmma<<<GRID_GEMM_N, 256, LDS128, stream>>>(hbf, HinT, nullptr, hnb, NN, NHID, 0);
        spmm_add_kernel<<<EDGES / 16, 128, 0, stream>>>(edge_row, edge_col, edge_w, hnb, g_i, EDGES);
        // candidate n
        gemm_bf16_wmma<<<GRID_GEMM_N, 256, LDS128, stream>>>(hbf, HhsT, nullptr, g_h, NN, NHID, 0);
        gemm_bf16_wmma<<<GRID_GEMM_N, 256, LDS128, stream>>>(hbf, HhnT, nullptr, hnb, NN, NHID, 0);
        spmm_add_kernel<<<EDGES / 16, 128, 0, stream>>>(edge_row, edge_col, edge_w, hnb, g_h, EDGES);
        // gate combine
        gru_update_kernel<<<EW_GRID, 256, 0, stream>>>(in_r, in_i, in_n, g_r, g_i, g_h, h, NH);
    }

    // 5) post-GRU MLP (x1->g_r, x2->g_i, x3->g_h, bf16 staging in hbf/xbf)
    conv_bf16_kernel<<<4096, 256, 0, stream>>>(h, hbf, NH);
    gemm_bf16_wmma<<<GRID_GEMM_N, 256, LDS128, stream>>>(hbf, W1T, b1, g_r, NN, NHID, 1);
    conv_bf16_kernel<<<4096, 256, 0, stream>>>(g_r, xbf, NH);
    gemm_bf16_wmma<<<GRID_GEMM_N, 256, LDS128, stream>>>(xbf, W2T, b2, g_i, NN, NHID, 1);
    conv_bf16_kernel<<<4096, 256, 0, stream>>>(g_i, xbf, NH);
    gemm_bf16_wmma<<<GRID_GEMM_N, 256, LDS128, stream>>>(xbf, W3T, b3, g_h, NN, NHID, 1);

    // 6) layernorm -> z (second output region) + z_bf16
    layernorm_kernel<<<(NN + 7) / 8, 256, 0, stream>>>(g_h, gamma, beta, out_z, zbf, NN);

    // 7) decoder on node pairs
    pair_gather_kernel<<<(NPAIR * 256 + 255) / 256, 256, 0, stream>>>(zbf, x_idx, y_idx, Ag, NPAIR);
    gemm_bf16_wmma<<<GRID_GEMM_P, 256, LDS256, stream>>>(Ag, Wd1T, bd1, d1, NPAIR, 2 * NHID, 1);
    fc2_sigmoid_kernel<<<(NPAIR + 7) / 8, 256, 0, stream>>>(d1, Wd2, bd2, out_d, NPAIR);
}
